// MultiHeadSelfAttention_2293512536233
// MI455X (gfx1250) — compile-verified, matrix operand order checked by reading, not run on hardware
//
#include <hip/hip_runtime.h>

// ---------------------------------------------------------------------------
// MultiHeadSelfAttention forward for MI455X (gfx1250, wave32, WMMA).
// B=2, S=2048, E=1024, H=16, D=64.  fp16 WMMA (f32 accum) + flash-attention.
// ---------------------------------------------------------------------------

typedef _Float16 half_t;
typedef __attribute__((ext_vector_type(16))) _Float16 v16h;
typedef __attribute__((ext_vector_type(8)))  _Float16 v8h;
typedef __attribute__((ext_vector_type(8)))  float    v8f;

#define S_LEN   2048
#define E_DIM   1024
#define H_NUM   16
#define D_DIM   64
#define M_ROWS  4096          // B*S

// Load a 16x32 f16 A-fragment (or 32x16 B-fragment) per the CDNA5 VGPR layout:
// lane lr (=lane&15) owns row/col lr; VGPRs 0..3 hold K = kbase + hl*8 + 0..7,
// VGPRs 4..7 hold K = kbase + 16 + hl*8 + 0..7 (hl = lane>>4).
// Caller passes p = rowptr + kbase + hl*8; we read p[0..7] and p[16..23].
__device__ __forceinline__ v16h load_frag(const half_t* p) {
  v8h lo = *(const v8h*)(p);
  v8h hi = *(const v8h*)(p + 16);
  v16h r;
#pragma unroll
  for (int i = 0; i < 8; ++i) { r[i] = lo[i]; r[i + 8] = hi[i]; }
  return r;
}

__device__ __forceinline__ v8f wmma_f16(v16h a, v16h b, v8f c) {
  return __builtin_amdgcn_wmma_f32_16x16x32_f16(false, a, false, b,
                                                (short)0, c, false, false);
}

// ---------------------------------------------------------------------------
// fp32 -> fp16 elementwise convert
// ---------------------------------------------------------------------------
__global__ void convert_f32_to_f16(const float* __restrict__ src,
                                   half_t* __restrict__ dst, int n) {
  int i = blockIdx.x * blockDim.x + threadIdx.x;
  if (i < n) dst[i] = (half_t)src[i];
}

// ---------------------------------------------------------------------------
// W[k][n] (1024x1024 f32, row-major) -> Wt[n][k] (f16)  via LDS tile
// ---------------------------------------------------------------------------
__global__ __launch_bounds__(256)
void transpose_convert(const float* __restrict__ W, half_t* __restrict__ Wt) {
  __shared__ float tile[32][33];
  const int tn = blockIdx.x & 31;      // n-tile
  const int tk = blockIdx.x >> 5;      // k-tile
  const int tx = threadIdx.x;          // 0..31
  const int ty = threadIdx.y;          // 0..7
#pragma unroll
  for (int j = 0; j < 32; j += 8)
    tile[ty + j][tx] = W[(tk * 32 + ty + j) * E_DIM + tn * 32 + tx];
  __syncthreads();
#pragma unroll
  for (int j = 0; j < 32; j += 8)
    Wt[(tn * 32 + ty + j) * E_DIM + tk * 32 + tx] = (half_t)tile[tx][ty + j];
}

// ---------------------------------------------------------------------------
// GEMM: Y[m][n] = X[m][:] . W[:][n] + bias[n],  M=4096, N=1024, K=1024.
// X f16 row-major (ld=1024), Wt f16 = W transposed (Wt[n][k], ld=1024).
// One wave computes a 32x64 tile (2x4 WMMA C-tiles), k-loop fully unrolled
// with register double-buffering so next-iter loads overlap current WMMAs.
// mode 0: store Q as f16 [B,H,S,D]
// mode 1: store K as f16 [B,H,S,D]
// mode 2: store V as f16 [B,H,D,S]   (transposed for P.V B-fragments)
// mode 3: store f32 to out (final projection)
// ---------------------------------------------------------------------------
__global__ __launch_bounds__(256)
void gemm32x64(const half_t* __restrict__ X, const half_t* __restrict__ Wt,
               const float* __restrict__ bias, half_t* __restrict__ dstH,
               float* __restrict__ dstF, int mode) {
  const int lane = threadIdx.x & 31;
  const int wid  = threadIdx.x >> 5;
  const int w    = blockIdx.x * (blockDim.x >> 5) + wid;   // 0..2047
  const int lr   = lane & 15;
  const int hl   = lane >> 4;
  const int m0   = (w >> 4) << 5;     // 32-row block  (128 m-blocks)
  const int n0   = (w & 15) << 6;     // 64-col block  (16  n-blocks)

  v8f acc[2][4] = {{{}, {}, {}, {}}, {{}, {}, {}, {}}};
  const half_t* xrow = X + (m0 + lr) * E_DIM + hl * 8;
  const half_t* wrow = Wt + (n0 + lr) * E_DIM + hl * 8;

  v16h A[2][2], B[2][4];
#pragma unroll
  for (int i = 0; i < 2; ++i) A[0][i] = load_frag(xrow + i * 16 * E_DIM);
#pragma unroll
  for (int t = 0; t < 4; ++t) B[0][t] = load_frag(wrow + t * 16 * E_DIM);

  constexpr int KSTEPS = E_DIM / 32;  // 32
#pragma unroll
  for (int kk = 0; kk < KSTEPS; ++kk) {
    const int cur = kk & 1, nxt = cur ^ 1;
    if (kk + 1 < KSTEPS) {
      const int k = (kk + 1) * 32;
#pragma unroll
      for (int i = 0; i < 2; ++i) A[nxt][i] = load_frag(xrow + i * 16 * E_DIM + k);
#pragma unroll
      for (int t = 0; t < 4; ++t) B[nxt][t] = load_frag(wrow + t * 16 * E_DIM + k);
    }
#pragma unroll
    for (int i = 0; i < 2; ++i)
#pragma unroll
      for (int t = 0; t < 4; ++t)
        acc[i][t] = wmma_f16(A[cur][i], B[cur][t], acc[i][t]);
  }

#pragma unroll
  for (int i = 0; i < 2; ++i) {
#pragma unroll
    for (int t = 0; t < 4; ++t) {
      const int col = n0 + t * 16 + lr;
      const float bv = bias[col];
#pragma unroll
      for (int r = 0; r < 8; ++r) {
        const int row = m0 + i * 16 + r + hl * 8;  // token index
        const float v = acc[i][t][r] + bv;
        if (mode == 3) {
          dstF[row * E_DIM + col] = v;
        } else {
          const int b  = row >> 11;                // /2048
          const int s  = row & 2047;
          const int h  = col >> 6;                 // /64
          const int d  = col & 63;
          const int bh = b * H_NUM + h;
          if (mode == 2) dstH[(bh * D_DIM + d) * S_LEN + s] = (half_t)v;
          else           dstH[(bh * S_LEN + s) * D_DIM + d] = (half_t)v;
        }
      }
    }
  }
}

// ---------------------------------------------------------------------------
// Flash attention: one wave owns 16 queries x D=64 for one (b,h).
// Streams keys in blocks of 32: 4 score WMMAs + online softmax + 4 PV WMMAs.
// Q A-fragments hoisted out of the loop; V B-fragments loaded at loop top so
// their latency hides behind score WMMAs + softmax VALU work.
// Q,K: f16 [B,H,S,D]; Vt: f16 [B,H,D,S]; out: f16 [B,S,E] (concat layout).
// ---------------------------------------------------------------------------
__global__ __launch_bounds__(256)
void flash_attn(const half_t* __restrict__ Qh, const half_t* __restrict__ Kh,
                const half_t* __restrict__ Vth, const int* __restrict__ mask,
                half_t* __restrict__ attn_out) {
  __shared__ half_t plds[8][16 * 32];            // per-wave P staging tile

  const int lane = threadIdx.x & 31;
  const int wid  = threadIdx.x >> 5;
  const int w    = blockIdx.x * (blockDim.x >> 5) + wid;   // 0..4095
  const int bh   = w >> 7;                       // (b*16+h), 0..31
  const int q0   = (w & 127) << 4;               // query block base
  const int b    = bh >> 4;
  const int h    = bh & 15;
  const int lr   = lane & 15;
  const int hl   = lane >> 4;

  float mrow[8], lsum[8];
#pragma unroll
  for (int r = 0; r < 8; ++r) { mrow[r] = -3.0e38f; lsum[r] = 0.0f; }
  v8f o[4] = {{}, {}, {}, {}};

  const half_t* qbase = Qh + (bh * S_LEN + q0 + lr) * D_DIM + hl * 8;
  const half_t* kbase = Kh + (bh * S_LEN) * D_DIM;
  const half_t* vbase = Vth + (bh * D_DIM) * S_LEN;
  const int*    mptr  = mask + b * S_LEN;

  // Q fragments are invariant over the key loop.
  const v16h qa0 = load_frag(qbase);
  const v16h qa1 = load_frag(qbase + 32);

  for (int kb = 0; kb < S_LEN; kb += 32) {
    // ---- V fragments first: latency hidden by score WMMAs + softmax -------
    v16h vb[4];
#pragma unroll
    for (int t = 0; t < 4; ++t)
      vb[t] = load_frag(vbase + (t * 16 + lr) * S_LEN + kb + hl * 8);

    // ---- all four K fragments, then four score WMMAs ----------------------
    const half_t* kp = kbase + (kb + lr) * D_DIM + hl * 8;
    v16h k00 = load_frag(kp);
    v16h k01 = load_frag(kp + 32);
    v16h k10 = load_frag(kp + 16 * D_DIM);
    v16h k11 = load_frag(kp + 16 * D_DIM + 32);

    if (kb + 32 < S_LEN) {   // prefetch next key block (global_prefetch_b8)
      __builtin_prefetch(kbase + (kb + 32 + lr) * D_DIM, 0, 3);
      __builtin_prefetch(vbase + lr * S_LEN + kb + 32, 0, 3);
    }

    v8f s0 = {}, s1 = {};
    s0 = wmma_f16(qa0, k00, s0);
    s0 = wmma_f16(qa1, k01, s0);
    s1 = wmma_f16(qa0, k10, s1);
    s1 = wmma_f16(qa1, k11, s1);

    const float mb0 = (1.0f - (float)mptr[kb + lr])      * -1e9f;
    const float mb1 = (1.0f - (float)mptr[kb + 16 + lr]) * -1e9f;

    asm volatile("" ::: "memory");
    // ---- online softmax per row (row M = r + 8*hl, reductions within 16) ---
#pragma unroll
    for (int r = 0; r < 8; ++r) {
      float x0 = s0[r] * 0.125f + mb0;           // 1/sqrt(64)
      float x1 = s1[r] * 0.125f + mb1;
      float v = fmaxf(x0, x1);
      v = fmaxf(v, __shfl_xor(v, 1, 16));
      v = fmaxf(v, __shfl_xor(v, 2, 16));
      v = fmaxf(v, __shfl_xor(v, 4, 16));
      v = fmaxf(v, __shfl_xor(v, 8, 16));
      const float newm = fmaxf(mrow[r], v);
      const float corr = __expf(mrow[r] - newm);
      mrow[r] = newm;
      const float p0 = __expf(x0 - newm);
      const float p1 = __expf(x1 - newm);
      float rs = p0 + p1;
      rs += __shfl_xor(rs, 1, 16);
      rs += __shfl_xor(rs, 2, 16);
      rs += __shfl_xor(rs, 4, 16);
      rs += __shfl_xor(rs, 8, 16);
      lsum[r] = lsum[r] * corr + rs;
      o[0][r] *= corr; o[1][r] *= corr; o[2][r] *= corr; o[3][r] *= corr;
      // stage P (C-layout -> row-major 16x32) in this wave's LDS patch
      const int row = r + hl * 8;
      plds[wid][row * 32 + lr]      = (half_t)p0;
      plds[wid][row * 32 + 16 + lr] = (half_t)p1;
    }
    asm volatile("" ::: "memory");               // same-wave DS is in-order

    // ---- O += P (16x32) x V (32x64), V fragments already resident ---------
    v16h pa = load_frag(&plds[wid][lr * 32 + hl * 8]);
#pragma unroll
    for (int t = 0; t < 4; ++t)
      o[t] = wmma_f16(pa, vb[t], o[t]);
    asm volatile("" ::: "memory");
  }

  // ---- normalize and store concat [B,S,E] in f16 for final projection ----
#pragma unroll
  for (int t = 0; t < 4; ++t) {
#pragma unroll
    for (int r = 0; r < 8; ++r) {
      const int token = b * S_LEN + q0 + r + hl * 8;
      const int col   = h * D_DIM + t * 16 + lr;
      attn_out[token * E_DIM + col] = (half_t)(o[t][r] / lsum[r]);
    }
  }
}

// ---------------------------------------------------------------------------
// Launch
// inputs: x, mask, Wq, bq, Wk, bk, Wv, bv, Wo, bo
// ---------------------------------------------------------------------------
extern "C" void kernel_launch(void* const* d_in, const int* in_sizes, int n_in,
                              void* d_out, int out_size, void* d_ws, size_t ws_size,
                              hipStream_t stream) {
  const float* x    = (const float*)d_in[0];
  const int*   mask = (const int*)d_in[1];
  const float* Wq = (const float*)d_in[2]; const float* bq = (const float*)d_in[3];
  const float* Wk = (const float*)d_in[4]; const float* bk = (const float*)d_in[5];
  const float* Wv = (const float*)d_in[6]; const float* bv = (const float*)d_in[7];
  const float* Wo = (const float*)d_in[8]; const float* bo = (const float*)d_in[9];
  float* out = (float*)d_out;

  char* ws = (char*)d_ws;
  const size_t MB = 1u << 20;
  half_t* x_h    = (half_t*)(ws + 0 * MB);    // 4096x1024       (8 MiB)
  half_t* WtQ    = (half_t*)(ws + 8 * MB);    // 1024x1024       (2 MiB)
  half_t* WtK    = (half_t*)(ws + 10 * MB);
  half_t* WtV    = (half_t*)(ws + 12 * MB);
  half_t* WtO    = (half_t*)(ws + 14 * MB);
  half_t* Qh     = (half_t*)(ws + 16 * MB);   // [B,H,S,D]       (8 MiB)
  half_t* Kh     = (half_t*)(ws + 24 * MB);   // [B,H,S,D]
  half_t* Vth    = (half_t*)(ws + 32 * MB);   // [B,H,D,S]
  half_t* attn_h = (half_t*)(ws + 40 * MB);   // 4096x1024 concat

  // 1) converts / transposes
  const int nX = M_ROWS * E_DIM;
  convert_f32_to_f16<<<(nX + 255) / 256, 256, 0, stream>>>(x, x_h, nX);
  dim3 tb(32, 8);
  transpose_convert<<<1024, tb, 0, stream>>>(Wq, WtQ);
  transpose_convert<<<1024, tb, 0, stream>>>(Wk, WtK);
  transpose_convert<<<1024, tb, 0, stream>>>(Wv, WtV);
  transpose_convert<<<1024, tb, 0, stream>>>(Wo, WtO);

  // 2) projections: 2048 waves each (one 32x64 tile per wave)
  gemm32x64<<<256, 256, 0, stream>>>(x_h, WtQ, bq, Qh,  nullptr, 0);
  gemm32x64<<<256, 256, 0, stream>>>(x_h, WtK, bk, Kh,  nullptr, 1);
  gemm32x64<<<256, 256, 0, stream>>>(x_h, WtV, bv, Vth, nullptr, 2);

  // 3) flash attention: 4096 waves (16 queries x one head each)
  flash_attn<<<512, 256, 0, stream>>>(Qh, Kh, Vth, mask, attn_h);

  // 4) output projection (fp32 + bias to d_out)
  gemm32x64<<<256, 256, 0, stream>>>(attn_h, WtO, bo, nullptr, out, 3);
}
